// SoftGumbelVQ_65953517797733
// MI455X (gfx1250) — matrix operand order repaired and checked by reading, not compiled, not run on hardware
//
#include <hip/hip_runtime.h>
#include <hip/hip_bf16.h>

typedef __attribute__((ext_vector_type(16))) _Float16 v16h;
typedef __attribute__((ext_vector_type(8)))  float    v8f;
typedef unsigned int su4 __attribute__((ext_vector_type(4)));
typedef unsigned int su8 __attribute__((ext_vector_type(8)));
typedef _Float16 h16;

constexpr int D   = 128;   // embedding dim
constexpr int K   = 256;   // codebook size
constexpr int TM  = 64;    // rows per tile
constexpr int LGS = 257;   // padded logits row stride (bank-conflict-free)
constexpr int GRID = 1024; // persistent workgroups

// dynamic LDS byte offsets
constexpr int oB   = 0;                 // h16 [16][4][32][16] = 64 KB (B fragments, loaded once by TDM)
constexpr int oZn  = oB   + K*D*2;      // h16 [64][128]       = 16 KB
constexpr int oLog = oZn  + TM*D*2;     // f32 [64][257]       = 65792 B (also z fp32 stage)
constexpr int oPt  = oLog + TM*LGS*4;   // f32 [128] partials (norm / expsum)
constexpr int oPV  = oPt  + 128*4;      // f32 [128] max partials
constexpr int oPI  = oPV  + 128*4;      // i32 [128] argmax partials
constexpr int oInv = oPI  + 128*4;      // f32 [64] 1/||z||
constexpr int oM   = oInv + TM*4;       // f32 [64] row max
constexpr int oIS  = oM   + TM*4;       // f32 [64] 1/sumexp
constexpr int oIdx = oIS  + TM*4;       // i32 [64] argmax
constexpr int oAvg = oIdx + TM*4;       // f32 [256] per-WG avg_probs accumulator
constexpr int SMEM = oAvg + K*4;        // ~147.8 KB -> 2 WGs / WGP (320 KB)

// ---------------- prep: normalize codebook into WMMA-B lane layout ----------------
__global__ __launch_bounds__(256) void vq_prep(const float* __restrict__ emb,
                                               h16* __restrict__ Bws,
                                               float* __restrict__ avg) {
    __shared__ float sinv[K];
    const int t = threadIdx.x;
    avg[t] = 0.0f;  // zero avg_probs accumulator every call
    {
        const float4* e4 = (const float4*)(emb + (size_t)t * D);
        float s = 0.f;
        #pragma unroll
        for (int i = 0; i < D/4; ++i) {
            float4 v = e4[i];
            s += v.x*v.x + v.y*v.y + v.z*v.z + v.w*v.w;
        }
        sinv[t] = 1.0f / fmaxf(sqrtf(s), 1e-12f);
    }
    __syncthreads();
    // idx = ((nt*4+c)*32+lane)*16+h ; col = nt*16+(lane&15); kdim = 32c+(lane>=16?16:0)+h
    for (int e = t; e < K*D; e += 256) {
        int h  = e & 15;
        int ln = (e >> 4) & 31;
        int c  = (e >> 9) & 3;
        int nt = e >> 11;
        int col = nt*16 + (ln & 15);
        int kd  = c*32 + ((ln >> 4) << 4) + h;
        Bws[e] = (h16)(emb[col*D + kd] * sinv[col]);
    }
}

// ---------------- main: persistent tiles; TDM-staged B; WMMA logits; softmax ----------------
__global__ __launch_bounds__(128) void vq_main(const float* __restrict__ z,
                                               const float* __restrict__ emb,
                                               const float* __restrict__ lscale,
                                               const h16*  __restrict__ Bws,
                                               float* __restrict__ avg,
                                               float* __restrict__ out_zq,
                                               float* __restrict__ out_idx,
                                               int Ntot) {
    extern __shared__ __align__(16) char smem[];
    h16*   sB   = (h16*)  (smem + oB);
    h16*   sZn  = (h16*)  (smem + oZn);
    float* sLog = (float*)(smem + oLog);
    float* sPt  = (float*)(smem + oPt);
    float* sPV  = (float*)(smem + oPV);
    int*   sPI  = (int*)  (smem + oPI);
    float* sInv = (float*)(smem + oInv);
    float* sM   = (float*)(smem + oM);
    float* sIS  = (float*)(smem + oIS);
    int*   sIdx = (int*)  (smem + oIdx);
    float* sAvg = (float*)(smem + oAvg);

    const int t    = threadIdx.x;
    const int lane = t & 31;
    const int w    = t >> 5;
    const float scl = lscale[0];
    const int nTiles = Ntot / TM;

    // ---- one-shot: Tensor Data Mover DMA of the 64KB B-fragment buffer into LDS ----
    if (t == 0) {
        unsigned long long ga = (unsigned long long)(uintptr_t)Bws;
        unsigned int ldsa = (unsigned int)(uintptr_t)sB;     // low 32 bits = LDS byte offset
        su4 g0;
        g0[0] = 1u;                                          // count=1, user desc, load
        g0[1] = ldsa;                                        // lds_addr
        g0[2] = (unsigned int)ga;                            // global_addr[31:0]
        g0[3] = (unsigned int)((ga >> 32) & 0x01FFFFFFu) | 0x80000000u; // ga[56:32] | type=2
        su8 g1;
        g1[0] = 2u << 16;                                    // data_size=4B
        g1[1] = (8192u & 0xFFFFu) << 16;                     // tensor_dim0 = 8192 dwords (lo16)
        g1[2] = (8192u >> 16) | (1u << 16);                  // tensor_dim0 hi16 | tensor_dim1=1 lo16
        g1[3] = 8192u << 16;                                 // tensor_dim1 hi16=0 | tile_dim0=8192
        g1[4] = 1u;                                          // tile_dim1=1 | tile_dim2=0
        g1[5] = 8192u;                                       // tensor_dim0_stride lo32
        g1[6] = 0u;                                          // stride hi16 | dim1_stride lo16
        g1[7] = 0u;                                          // dim1_stride hi32
        asm volatile("tensor_load_to_lds %0, %1" :: "s"(g0), "s"(g1) : "memory");
    }

    // per-thread-owned avg accumulator slots (k = t, t+128): zero once
    sAvg[t] = 0.0f;
    sAvg[t + 128] = 0.0f;

    for (int ti = blockIdx.x; ti < nTiles; ti += GRID) {
        const int row0 = ti * TM;

        // prefetch next tile's z rows (global_prefetch_b8)
        {
            int tin = ti + GRID;
            if (tin < nTiles)
                __builtin_prefetch((const char*)z + (size_t)tin * TM * D * 4 + (size_t)t * 256, 0, 0);
        }

        // ---- load z tile (fp32 stage in logits region), per-row sum of squares ----
        {
            float* zst = sLog;                       // [64][128] fp32 staging
            const int r = t >> 1, hh = t & 1;
            const float4* gz = (const float4*)(z + (size_t)(row0 + r) * D + hh*64);
            float4* st = (float4*)(zst + r*D + hh*64);
            float s = 0.f;
            #pragma unroll
            for (int i = 0; i < 16; ++i) {
                float4 v = gz[i];
                st[i] = v;
                s += v.x*v.x + v.y*v.y + v.z*v.z + v.w*v.w;
            }
            sPt[t] = s;
        }
        __syncthreads();
        if (t < TM) {
            float s = sPt[2*t] + sPt[2*t + 1];
            sInv[t] = 1.0f / fmaxf(sqrtf(s), 1e-12f);
        }
        __syncthreads();
        {
            const float* zst = sLog;
            for (int e = t; e < TM*D; e += 128) {
                int r = e >> 7;
                sZn[e] = (h16)(zst[e] * sInv[r]);
            }
        }
        // first iteration: make sure the TDM B-load has landed before anyone enters WMMA
        if (ti == blockIdx.x && t == 0) __builtin_amdgcn_s_wait_tensorcnt(0);
        __syncthreads();

        // ---- WMMA: per wave, 16 rows x 256 cols, f32 accumulate, double-buffered B ----
        {
            const int mrow = lane & 15;
            const int side = lane >> 4;
            v16h a[4];
            #pragma unroll
            for (int c = 0; c < 4; ++c) {
                #pragma unroll
                for (int h = 0; h < 16; ++h) {
                    int kk = c*32 + side*16 + h;
                    a[c][h] = sZn[(w*16 + mrow)*D + kk];
                }
            }
            v16h b[2][4];
            #pragma unroll
            for (int c = 0; c < 4; ++c)
                b[0][c] = *(const v16h*)&sB[((0*4 + c)*32 + lane)*16];
            #pragma unroll
            for (int nt = 0; nt < 16; ++nt) {
                const int cur = nt & 1, nxt = cur ^ 1;
                if (nt < 15) {
                    #pragma unroll
                    for (int c = 0; c < 4; ++c)
                        b[nxt][c] = *(const v16h*)&sB[(((nt+1)*4 + c)*32 + lane)*16];
                }
                v8f acc = {};
                #pragma unroll
                for (int c = 0; c < 4; ++c)
                    acc = __builtin_amdgcn_wmma_f32_16x16x32_f16(
                              false, a[c], false, b[cur][c], (short)0, acc, false, false);
                #pragma unroll
                for (int r = 0; r < 8; ++r)
                    sLog[(w*16 + side*8 + r)*LGS + nt*16 + mrow] = acc[r] * scl;
            }
        }
        __syncthreads();

        // ---- per-row argmax + softmax stats, 2 threads per row ----
        {
            const int r = t & 63, hh = t >> 6;       // half: k in [128*hh, 128*hh+128)
            const float* Lr = sLog + r*LGS + hh*128;
            float m = -3.4e38f; int bi = 0;
            for (int k = 0; k < 128; ++k) { float v = Lr[k]; if (v > m) { m = v; bi = k; } }
            sPV[t] = m; sPI[t] = bi + hh*128;
        }
        __syncthreads();
        if (t < TM) {
            float v0 = sPV[t], v1 = sPV[t + 64];
            sM[t]   = fmaxf(v0, v1);
            sIdx[t] = (v0 >= v1) ? sPI[t] : sPI[t + 64];  // first-occurrence tie-break
        }
        __syncthreads();
        {
            const int r = t & 63, hh = t >> 6;
            const float* Lr = sLog + r*LGS + hh*128;
            const float m = sM[r];
            float s = 0.f;
            for (int k = 0; k < 128; ++k) s += __expf(Lr[k] - m);
            sPt[t] = s;
        }
        __syncthreads();
        if (t < TM) sIS[t] = 1.0f / (sPt[t] + sPt[t + 64]);
        __syncthreads();

        // ---- column-parallel avg_probs partials into per-WG LDS accumulator ----
        #pragma unroll
        for (int kk = 0; kk < 2; ++kk) {
            const int k = t + kk*128;
            float a = 0.f;
            for (int r = 0; r < TM; ++r)
                a += __expf(sLog[r*LGS + k] - sM[r]) * sIS[r];
            sAvg[k] += a;      // slot owned exclusively by this thread
        }

        // ---- outputs: indices (as float) + coalesced z_q gather ----
        if (t < TM) out_idx[row0 + t] = (float)sIdx[t];
        {
            const float4* e4 = (const float4*)emb;
            float4* o4 = (float4*)out_zq;
            for (int e = t; e < TM*(D/4); e += 128) {
                int r = e >> 5, q = e & 31;
                o4[(size_t)(row0 + r)*(D/4) + q] = e4[(size_t)sIdx[r]*(D/4) + q];
            }
        }
        __syncthreads();   // protect sLog/sM/sIS/sIdx before next tile overwrites
    }

    // ---- flush per-WG avg_probs (one atomic per column per WG) ----
    {
        const float invN = 1.0f / (float)Ntot;
        atomicAdd(&avg[t],       sAvg[t]       * invN);
        atomicAdd(&avg[t + 128], sAvg[t + 128] * invN);
    }
}

// ---------------- finalize: perplexity ----------------
__global__ __launch_bounds__(256) void vq_fin(const float* __restrict__ avg,
                                              float* __restrict__ out_pp) {
    __shared__ float red[256];
    const int t = threadIdx.x;
    float p = avg[t];
    red[t] = p * logf(p + 1e-10f);
    __syncthreads();
    for (int s = 128; s > 0; s >>= 1) {
        if (t < s) red[t] += red[t + s];
        __syncthreads();
    }
    if (t == 0) out_pp[0] = expf(-red[0]);
}

extern "C" void kernel_launch(void* const* d_in, const int* in_sizes, int n_in,
                              void* d_out, int out_size, void* d_ws, size_t ws_size,
                              hipStream_t stream) {
    const float* z   = (const float*)d_in[0];   // [N,128]
    const float* emb = (const float*)d_in[1];   // [256,128]
    const float* lsc = (const float*)d_in[2];   // [1]
    const int N = in_sizes[0] / D;              // 524288

    float* out    = (float*)d_out;
    float* out_zq = out;                         // N*128
    float* out_ix = out + (size_t)N * D;         // N
    float* out_pp = out + (size_t)N * D + N;     // 1

    h16*   Bws = (h16*)d_ws;                     // 64 KB B-fragment buffer
    float* avg = (float*)((char*)d_ws + K*D*2);  // 1 KB avg_probs accumulator

    (void)hipFuncSetAttribute((const void*)vq_main,
                              hipFuncAttributeMaxDynamicSharedMemorySize, SMEM);

    vq_prep<<<1, 256, 0, stream>>>(emb, Bws, avg);
    vq_main<<<GRID, 128, SMEM, stream>>>(z, emb, lsc, Bws, avg, out_zq, out_ix, N);
    vq_fin<<<1, 256, 0, stream>>>(avg, out_pp);
}
